// QWindowAttention_6004364280064
// MI455X (gfx1250) — matrix operand order repaired and checked by reading, not compiled, not run on hardware
//
#include <hip/hip_runtime.h>
#include <hip/hip_bf16.h>

// ---------------------------------------------------------------------------
// Swin window attention, fused, WMMA f16 (fp32 accumulate) for gfx1250.
// One 256-thread block (8 wave32) per window. 49 tokens padded to 64.
// ---------------------------------------------------------------------------

typedef _Float16 v16h __attribute__((ext_vector_type(16)));
typedef _Float16 v8h  __attribute__((ext_vector_type(8)));
typedef _Float16 v4h  __attribute__((ext_vector_type(4)));
typedef float    v8f  __attribute__((ext_vector_type(8)));

#define WINSZ 7
#define NTOK  49
#define DIMC  128
#define HEADS 4
#define HD    32
#define NPAD  64
#define NWIN  64
#define SCALE 0.17677669529663687f   // 1/sqrt(32)

union V16 { v16h v; v8h h[2]; };

// A fragment (16x32 f16) from a row-major f16 buffer, row stride rs elements.
// Lane l holds row (row0 + l%16); K order matches B: elems i = K 16g+i
// (g = l/16) -> two 16B chunks at +16g and +16g+8.
__device__ __forceinline__ v16h load_a_frag(const _Float16* buf, int rs,
                                            int row0, int k0, int lane) {
  const int m = lane & 15, g = lane >> 4;
  const _Float16* p = buf + (row0 + m) * rs + k0 + g * 16;
  V16 u;
  u.h[0] = *(const v8h*)p;
  u.h[1] = *(const v8h*)(p + 8);
  return u.v;
}

// B fragment (32x16 f16): lane l = column l%16, elems i = K 16*(l/16)+i.
// p must point at element (k0 + 16*(l/16), col) with K contiguous in memory.
__device__ __forceinline__ v16h load_b_frag(const _Float16* p) {
  V16 u;
  u.h[0] = *(const v8h*)p;
  u.h[1] = *(const v8h*)(p + 8);
  return u.v;
}

__device__ __forceinline__ v8f wmma_f16(v16h a, v16h b, v8f c) {
  return __builtin_amdgcn_wmma_f32_16x16x32_f16(false, a, false, b,
                                                (short)0, c, false, false);
}

// ---------------------------------------------------------------------------
// Prep kernel 1: convert w_qkv (128x384) and w_proj (128x128) fp32 -> f16 in
// WMMA B-fragment order: frag(kt,nt) is 512 halves; half index = lane*16 + i
// with lane = (K%32/16)*16 + n%16, i = K%16.
// ---------------------------------------------------------------------------
__global__ void prep_weights(const float* __restrict__ w_qkv,
                             const float* __restrict__ w_proj,
                             _Float16* __restrict__ wq,
                             _Float16* __restrict__ wp) {
  int idx = blockIdx.x * 256 + threadIdx.x;
  if (idx < DIMC * 384) {
    int k = idx / 384, n = idx % 384;
    int kt = k >> 5, nt = n >> 4, kk = k & 31, nn = n & 15;
    int lane = ((kk >> 4) << 4) | nn, i = kk & 15;
    wq[((kt * 24 + nt) << 9) + lane * 16 + i] = (_Float16)w_qkv[idx];
  } else if (idx < DIMC * 384 + DIMC * DIMC) {
    int j = idx - DIMC * 384;
    int k = j >> 7, n = j & 127;
    int kt = k >> 5, nt = n >> 4, kk = k & 31, nn = n & 15;
    int lane = ((kk >> 4) << 4) | nn, i = kk & 15;
    wp[((kt * 8 + nt) << 9) + lane * 16 + i] = (_Float16)w_proj[j];
  }
}

// ---------------------------------------------------------------------------
// Prep kernel 2: expand relative-position bias table to [H][64][64] fp32,
// zero outside the 49x49 valid region. REL_IDX recomputed on device.
// ---------------------------------------------------------------------------
__global__ void prep_bias(const float* __restrict__ bias_table,
                          float* __restrict__ bias_full) {
  int idx = blockIdx.x * 256 + threadIdx.x;   // 4*64*64 = 16384
  if (idx >= HEADS * NPAD * NPAD) return;
  int h = idx >> 12;
  int i = (idx >> 6) & 63;
  int j = idx & 63;
  float v = 0.0f;
  if (i < NTOK && j < NTOK) {
    int yi = i / WINSZ, xi = i % WINSZ, yj = j / WINSZ, xj = j % WINSZ;
    int rel = (yi - yj + WINSZ - 1) * (2 * WINSZ - 1) + (xi - xj + WINSZ - 1);
    v = bias_table[rel * HEADS + h];
  }
  bias_full[idx] = v;
}

// ---------------------------------------------------------------------------
// Main fused attention kernel. 64 KB LDS, phase-reused.
// ---------------------------------------------------------------------------
struct __align__(16) Smem {
  union { _Float16 xs[NPAD * DIMC]; _Float16 os[NPAD * DIMC]; };     // 16 KB
  union { _Float16 qs[NPAD * DIMC]; _Float16 pbuf[8 * 16 * NPAD]; }; // 16 KB
  _Float16 ks[NPAD * DIMC];                                          // 16 KB
  _Float16 vsT[DIMC * NPAD];   // V transposed: [feat][token]        // 16 KB
};

__global__ __launch_bounds__(256) void swin_attn(
    const float* __restrict__ x, const float* __restrict__ mask,
    const float* __restrict__ b_qkv, const float* __restrict__ b_proj,
    const _Float16* __restrict__ wq, const _Float16* __restrict__ wp,
    const float* __restrict__ bias_full, float* __restrict__ out) {
  __shared__ Smem sm;
  const int b    = blockIdx.x;
  const int widx = b & (NWIN - 1);
  const int tid  = threadIdx.x;
  // wave id is uniform within a wave: pin it to an SGPR so all per-wave tile
  // indices become scalar control flow instead of exec-mask divergence.
  const int wave = __builtin_amdgcn_readfirstlane(tid >> 5);
  const int lane = tid & 31;
  const int n16  = lane & 15, g = lane >> 4;

  __builtin_prefetch(wq, 0, 0);
  __builtin_prefetch(wp, 0, 0);

  // -------- Phase 1: load x window (49x128 fp32) -> xs f16, zero-pad rows.
  // float4 global loads -> packed f16 -> 8B LDS stores.
  {
    const float4* xg4 = (const float4*)(x + (size_t)b * NTOK * DIMC);
    for (int i = tid; i < (NPAD * DIMC) / 4; i += 256) {
      int r = i >> 5;                       // row = (i*4) / 128
      float4 t;
      if (r < NTOK) t = xg4[i];
      else          t = make_float4(0.f, 0.f, 0.f, 0.f);
      v4h hv = {(_Float16)t.x, (_Float16)t.y, (_Float16)t.z, (_Float16)t.w};
      *(v4h*)(sm.xs + i * 4) = hv;
    }
  }
  __syncthreads();

  // -------- Phase 2: QKV GEMM (64x128 @ 128x384), 96 tiles, 12 per wave
  for (int it = 0; it < 12; ++it) {
    int t = wave * 12 + it;                 // scalar
    int mt = t / 24, nt = t % 24;           // scalar
    v8f c = {};
#pragma unroll
    for (int kt = 0; kt < 4; ++kt) {
      v16h a  = load_a_frag(sm.xs, DIMC, mt * 16, kt * 32, lane);
      v16h bf = load_b_frag(wq + ((kt * 24 + nt) << 9) + lane * 16);
      c = wmma_f16(a, bf, c);
    }
    float bq = b_qkv[nt * 16 + n16];
    if (nt < 8) {                      // Q, row-major [tok][feat]
#pragma unroll
      for (int r = 0; r < 8; ++r)
        sm.qs[(mt * 16 + r + 8 * g) * DIMC + nt * 16 + n16] =
            (_Float16)(c[r] + bq);
    } else if (nt < 16) {              // K, row-major [tok][feat]
#pragma unroll
      for (int r = 0; r < 8; ++r)
        sm.ks[(mt * 16 + r + 8 * g) * DIMC + (nt - 8) * 16 + n16] =
            (_Float16)(c[r] + bq);
    } else {                           // V, transposed [feat][tok]
#pragma unroll
      for (int r = 0; r < 8; ++r)
        sm.vsT[((nt - 16) * 16 + n16) * NPAD + mt * 16 + r + 8 * g] =
            (_Float16)(c[r] + bq);
    }
  }
  __syncthreads();

  // -------- Phase 2.5: preload this wave's Q A-fragments (2 tasks), then
  // the qs region is recycled as per-wave P buffers.
  v16h aq0, aq1;
  {
    int t0 = wave, t1 = wave + 8;                 // task = h*4 + mt
    aq0 = load_a_frag(sm.qs, DIMC, (t0 & 3) * 16, (t0 >> 2) * HD, lane);
    aq1 = load_a_frag(sm.qs, DIMC, (t1 & 3) * 16, (t1 >> 2) * HD, lane);
  }
  __syncthreads();

  _Float16* P = sm.pbuf + wave * (16 * NPAD);
  const float* maskw = mask + (size_t)widx * NTOK * NTOK;

  // -------- Phase 3: per-task S = Q K^T, softmax, O = P V
  for (int s = 0; s < 2; ++s) {
    int task = wave + s * 8;                 // scalar
    int h = task >> 2, mt = task & 3;        // scalar
    v16h aq = s ? aq1 : aq0;

    v8f sa[4];
#pragma unroll
    for (int j = 0; j < 4; ++j) {      // S tile: 16 queries x 16 keys, K=32
      v16h bk = load_b_frag(sm.ks + (j * 16 + n16) * DIMC + h * HD + g * 16);
      v8f z = {};
      sa[j] = wmma_f16(aq, bk, z);
    }
    // scale + rel-pos bias + window mask; kill padded columns
    const float* biash = bias_full + h * NPAD * NPAD;
#pragma unroll
    for (int j = 0; j < 4; ++j) {
      int col = j * 16 + n16;
#pragma unroll
      for (int r = 0; r < 8; ++r) {
        int row = mt * 16 + r + 8 * g;
        float v = sa[j][r] * SCALE + biash[row * NPAD + col];
        if (col < NTOK) {
          if (row < NTOK) v += maskw[row * NTOK + col];
        } else {
          v = -1e30f;
        }
        sa[j][r] = v;
      }
    }
    // softmax over 64 cols; each row lives in 16 lanes of one half-wave
#pragma unroll
    for (int r = 0; r < 8; ++r) {
      float m = fmaxf(fmaxf(sa[0][r], sa[1][r]), fmaxf(sa[2][r], sa[3][r]));
#pragma unroll
      for (int off = 1; off < 16; off <<= 1)
        m = fmaxf(m, __shfl_xor(m, off, 32));
      float sum = 0.0f;
#pragma unroll
      for (int j = 0; j < 4; ++j) {
        sa[j][r] = __expf(sa[j][r] - m);
        sum += sa[j][r];
      }
#pragma unroll
      for (int off = 1; off < 16; off <<= 1)
        sum += __shfl_xor(sum, off, 32);
      float inv = 1.0f / sum;
#pragma unroll
      for (int j = 0; j < 4; ++j) sa[j][r] *= inv;
    }
    // write P (16x64 f16, row-major) to this wave's slot
#pragma unroll
    for (int j = 0; j < 4; ++j)
#pragma unroll
      for (int r = 0; r < 8; ++r)
        P[(r + 8 * g) * NPAD + j * 16 + n16] = (_Float16)sa[j][r];
    asm volatile("s_wait_dscnt 0" ::: "memory");

    // O strip (16 x 32) = P (16x64) @ V_head (64x32)
    v8f o0 = {}, o1 = {};
#pragma unroll
    for (int kt = 0; kt < 2; ++kt) {
      v16h ap  = load_a_frag(P, NPAD, 0, kt * 32, lane);
      v16h bv0 = load_b_frag(sm.vsT + (h * HD + n16) * NPAD + kt * 32 + g * 16);
      v16h bv1 = load_b_frag(sm.vsT + (h * HD + 16 + n16) * NPAD + kt * 32 + g * 16);
      o0 = wmma_f16(ap, bv0, o0);
      o1 = wmma_f16(ap, bv1, o1);
    }
#pragma unroll
    for (int r = 0; r < 8; ++r) {
      int row = mt * 16 + r + 8 * g;
      sm.os[row * DIMC + h * HD + n16]      = (_Float16)o0[r];
      sm.os[row * DIMC + h * HD + 16 + n16] = (_Float16)o1[r];
    }
  }
  __syncthreads();

  // -------- Phase 4: projection (64x128 @ 128x128) + b_proj, store rows<49
  float* og = out + (size_t)b * NTOK * DIMC;
  for (int it = 0; it < 4; ++it) {
    int t = wave * 4 + it;                   // scalar
    int mt = t >> 3, nt = t & 7;             // scalar
    v8f c = {};
#pragma unroll
    for (int kt = 0; kt < 4; ++kt) {
      v16h a  = load_a_frag(sm.os, DIMC, mt * 16, kt * 32, lane);
      v16h bf = load_b_frag(wp + ((kt * 8 + nt) << 9) + lane * 16);
      c = wmma_f16(a, bf, c);
    }
    float bp = b_proj[nt * 16 + n16];
#pragma unroll
    for (int r = 0; r < 8; ++r) {
      int row = mt * 16 + r + 8 * g;
      if (row < NTOK) og[row * DIMC + nt * 16 + n16] = c[r] + bp;
    }
  }
}

// ---------------------------------------------------------------------------
// Launch. d_ws layout: [0,96K) wqkv f16 frags | [96K,128K) wproj f16 frags |
// [128K,192K) bias_full fp32.
// ---------------------------------------------------------------------------
extern "C" void kernel_launch(void* const* d_in, const int* in_sizes, int n_in,
                              void* d_out, int out_size, void* d_ws, size_t ws_size,
                              hipStream_t stream) {
  const float* x          = (const float*)d_in[0];
  const float* mask       = (const float*)d_in[1];
  const float* w_qkv      = (const float*)d_in[2];
  const float* b_qkv      = (const float*)d_in[3];
  const float* w_proj     = (const float*)d_in[4];
  const float* b_proj     = (const float*)d_in[5];
  const float* bias_table = (const float*)d_in[6];

  _Float16* wq        = (_Float16*)d_ws;
  _Float16* wp        = (_Float16*)((char*)d_ws + 98304);
  float*    bias_full = (float*)((char*)d_ws + 131072);

  prep_weights<<<256, 256, 0, stream>>>(w_qkv, w_proj, wq, wp);
  prep_bias<<<64, 256, 0, stream>>>(bias_table, bias_full);

  int nwin_total = in_sizes[0] / (NTOK * DIMC);   // 8192
  swin_attn<<<nwin_total, 256, 0, stream>>>(x, mask, b_qkv, b_proj,
                                            wq, wp, bias_full, (float*)d_out);
}
